// GATLayer_88931592830991
// MI455X (gfx1250) — compile-verified, matrix operand order checked by reading, not run on hardware
//
#include <hip/hip_runtime.h>
#include <hip/hip_bf16.h>

#define NROWS 10000
#define INFEAT 512
#define OUTF 64
#define NPAD 10016              // hT columns padded to a multiple of 32
#define ALPHA 0.2f
#define NEGINF -9.0e15f

typedef __attribute__((ext_vector_type(16))) __bf16 v16bf;
typedef __attribute__((ext_vector_type(8)))  float  v8f;
typedef __attribute__((ext_vector_type(4)))  float  v4f;   // clang-native: OK for nontemporal builtins

union ABPack {
    v16bf v;
    unsigned short u[16];
    uint4 q[2];
};

__device__ __forceinline__ unsigned short f2bf(float f) {
    unsigned int x = __float_as_uint(f);
    x += 0x7fffu + ((x >> 16) & 1u);   // round-to-nearest-even
    return (unsigned short)(x >> 16);
}

__device__ __forceinline__ v8f wmma_bf16(const ABPack& A, const ABPack& B, v8f C) {
    return __builtin_amdgcn_wmma_f32_16x16x32_bf16(
        /*neg_a=*/false, A.v, /*neg_b=*/false, B.v,
        /*c_mod=*/(short)0, C, /*reuse_a=*/false, /*reuse_b=*/false);
}

// ---------------- Kernel 0: prep (W -> WT bf16, zero hT pad) ----------------
__global__ void k_prep(const float* __restrict__ W,
                       unsigned short* __restrict__ WT,
                       unsigned short* __restrict__ hT) {
    const int tid = threadIdx.x;
    for (int i = tid; i < OUTF * INFEAT; i += blockDim.x) {
        int f = i >> 9;            // / 512
        int k = i & 511;
        WT[i] = f2bf(W[k * OUTF + f]);
    }
    for (int i = tid; i < OUTF * (NPAD - NROWS); i += blockDim.x) {
        int f = i >> 4;            // / 16
        int c = i & 15;
        hT[(size_t)f * NPAD + NROWS + c] = 0;
    }
}

// ---------------- Kernel 1: h = x@W (WMMA bf16), s_self/s_neigh, hT ---------
__global__ void __launch_bounds__(256) k_hidden(
        const float* __restrict__ x, const unsigned short* __restrict__ WT,
        const float* __restrict__ a_self, const float* __restrict__ a_nei,
        unsigned short* __restrict__ hT,
        float* __restrict__ s_self, float* __restrict__ s_nei) {
    const int wave = blockIdx.x * (blockDim.x >> 5) + (threadIdx.x >> 5);
    if (wave * 16 >= NROWS) return;          // wave-uniform exit
    const int rb   = wave * 16;
    const int lane = threadIdx.x & 31;
    const int hi   = lane >> 4;              // 0: K{0-7,16-23}, 1: K{8-15,24-31}
    const int r    = lane & 15;              // A: row-in-tile, B: feature col

    const float* xrow = x + (size_t)(rb + r) * INFEAT;

    v8f C[4];
    #pragma unroll
    for (int t = 0; t < 4; ++t)
        #pragma unroll
        for (int i = 0; i < 8; ++i) C[t][i] = 0.f;

    for (int kb = 0; kb < INFEAT; kb += 32) {
        const int k0 = kb + hi * 8;
        v4f xa = *(const v4f*)(xrow + k0);
        v4f xb = *(const v4f*)(xrow + k0 + 4);
        v4f xc = *(const v4f*)(xrow + k0 + 16);
        v4f xd = *(const v4f*)(xrow + k0 + 20);
        float fx[16] = {xa.x, xa.y, xa.z, xa.w, xb.x, xb.y, xb.z, xb.w,
                        xc.x, xc.y, xc.z, xc.w, xd.x, xd.y, xd.z, xd.w};
        ABPack A;
        #pragma unroll
        for (int j = 0; j < 16; ++j) A.u[j] = f2bf(fx[j]);

        #pragma unroll
        for (int t = 0; t < 4; ++t) {
            const unsigned short* bp = WT + (t * 16 + r) * INFEAT + k0;
            ABPack B;
            B.q[0] = *(const uint4*)bp;
            B.q[1] = *(const uint4*)(bp + 16);
            C[t] = wmma_bf16(A, B, C[t]);
        }
    }

    // attention projections: per-row dot(h, a_*) with cross-lane reduction
    float avs[4], avn[4];
    #pragma unroll
    for (int t = 0; t < 4; ++t) {
        avs[t] = a_self[t * 16 + r];
        avn[t] = a_nei [t * 16 + r];
    }
    #pragma unroll
    for (int v = 0; v < 8; ++v) {
        float ps = 0.f, pn = 0.f;
        #pragma unroll
        for (int t = 0; t < 4; ++t) {
            float hv = C[t][v];
            ps += hv * avs[t];
            pn += hv * avn[t];
        }
        #pragma unroll
        for (int m = 1; m < 16; m <<= 1) {   // reduce within each 16-lane half
            ps += __shfl_xor(ps, m, 32);
            pn += __shfl_xor(pn, m, 32);
        }
        if (r == 0) {                        // rows 0-7 via lane 0, 8-15 via lane 16
            s_self[rb + hi * 8 + v] = ps;
            s_nei [rb + hi * 8 + v] = pn;
        }
    }

    // hT[feat][row] in bf16, 8 consecutive rows packed into one 16B store
    #pragma unroll
    for (int t = 0; t < 4; ++t) {
        ABPack P;
        #pragma unroll
        for (int v = 0; v < 8; ++v) P.u[v] = f2bf(C[t][v]);
        *(uint4*)(hT + (size_t)(t * 16 + r) * NPAD + rb + hi * 8) = P.q[0];
    }
}

// ---------------- Kernel 2: fused masked-softmax attention @ h --------------
__global__ void __launch_bounds__(128) k_attn(
        const float* __restrict__ adj, const float* __restrict__ Mm,
        const float* __restrict__ s_self, const float* __restrict__ s_nei,
        const unsigned short* __restrict__ hT, float* __restrict__ out) {
    __shared__ float lsC[4][4][8][32];   // [wave][tile][vgpr][lane] = 16 KB
    __shared__ float lsL[4][32];

    const int w    = threadIdx.x >> 5;
    const int lane = threadIdx.x & 31;
    const int hi   = lane >> 4;
    const int r    = lane & 15;
    const int row_base = blockIdx.x * 16;
    const int row  = row_base + r;

    const float  ss   = s_self[row];
    const float* Mrow = Mm  + (size_t)row * NROWS;
    const float* Grow = adj + (size_t)row * NROWS;

    v8f C[4];
    #pragma unroll
    for (int t = 0; t < 4; ++t)
        #pragma unroll
        for (int i = 0; i < 8; ++i) C[t][i] = 0.f;
    float lsum = 0.f;

    // main loop: full 32-col tiles, 4 waves interleaved by 128 columns
    for (int cb = w * 32; cb + 32 <= NROWS; cb += 128) {
        const int c0 = cb + hi * 8;
        // stream M/adj non-temporally (no reuse; keep L2 for hT)
        v4f m0 = __builtin_nontemporal_load((const v4f*)(Mrow + c0));
        v4f m1 = __builtin_nontemporal_load((const v4f*)(Mrow + c0 + 4));
        v4f m2 = __builtin_nontemporal_load((const v4f*)(Mrow + c0 + 16));
        v4f m3 = __builtin_nontemporal_load((const v4f*)(Mrow + c0 + 20));
        v4f g0 = __builtin_nontemporal_load((const v4f*)(Grow + c0));
        v4f g1 = __builtin_nontemporal_load((const v4f*)(Grow + c0 + 4));
        v4f g2 = __builtin_nontemporal_load((const v4f*)(Grow + c0 + 16));
        v4f g3 = __builtin_nontemporal_load((const v4f*)(Grow + c0 + 20));
        v4f n0 = *(const v4f*)(s_nei + c0);
        v4f n1 = *(const v4f*)(s_nei + c0 + 4);
        v4f n2 = *(const v4f*)(s_nei + c0 + 16);
        v4f n3 = *(const v4f*)(s_nei + c0 + 20);

        float fm[16] = {m0.x,m0.y,m0.z,m0.w, m1.x,m1.y,m1.z,m1.w,
                        m2.x,m2.y,m2.z,m2.w, m3.x,m3.y,m3.z,m3.w};
        float fg[16] = {g0.x,g0.y,g0.z,g0.w, g1.x,g1.y,g1.z,g1.w,
                        g2.x,g2.y,g2.z,g2.w, g3.x,g3.y,g3.z,g3.w};
        float fn[16] = {n0.x,n0.y,n0.z,n0.w, n1.x,n1.y,n1.z,n1.w,
                        n2.x,n2.y,n2.z,n2.w, n3.x,n3.y,n3.z,n3.w};

        ABPack A;
        #pragma unroll
        for (int j = 0; j < 16; ++j) {
            float t = (ss + fn[j]) * fm[j];
            t = fmaxf(t, ALPHA * t);            // leaky relu (alpha < 1)
            t = (fg[j] > 0.f) ? t : NEGINF;     // mask; exp -> exactly 0
            float p = __expf(t);
            lsum += p;
            A.u[j] = f2bf(p);
        }
        #pragma unroll
        for (int t2 = 0; t2 < 4; ++t2) {
            const unsigned short* bp = hT + (size_t)(t2 * 16 + r) * NPAD + c0;
            ABPack B;
            B.q[0] = *(const uint4*)bp;
            B.q[1] = *(const uint4*)(bp + 16);
            C[t2] = wmma_bf16(A, B, C[t2]);
        }
    }

    // tail: columns 9984..9999 (one half tile), wave 0 only (wave-uniform)
    if (w == 0) {
        const int c0 = 9984 + hi * 8;           // 8 valid cols per half-wave
        v4f m0 = __builtin_nontemporal_load((const v4f*)(Mrow + c0));
        v4f m1 = __builtin_nontemporal_load((const v4f*)(Mrow + c0 + 4));
        v4f g0 = __builtin_nontemporal_load((const v4f*)(Grow + c0));
        v4f g1 = __builtin_nontemporal_load((const v4f*)(Grow + c0 + 4));
        v4f n0 = *(const v4f*)(s_nei + c0);
        v4f n1 = *(const v4f*)(s_nei + c0 + 4);
        float fm[8] = {m0.x,m0.y,m0.z,m0.w, m1.x,m1.y,m1.z,m1.w};
        float fg[8] = {g0.x,g0.y,g0.z,g0.w, g1.x,g1.y,g1.z,g1.w};
        float fn[8] = {n0.x,n0.y,n0.z,n0.w, n1.x,n1.y,n1.z,n1.w};
        ABPack A;
        #pragma unroll
        for (int j = 0; j < 8; ++j) {
            float t = (ss + fn[j]) * fm[j];
            t = fmaxf(t, ALPHA * t);
            t = (fg[j] > 0.f) ? t : NEGINF;
            float p = __expf(t);
            lsum += p;
            A.u[j] = f2bf(p);
        }
        #pragma unroll
        for (int j = 8; j < 16; ++j) A.u[j] = 0;   // cols >= N: zero weight
        #pragma unroll
        for (int t2 = 0; t2 < 4; ++t2) {
            const unsigned short* bp = hT + (size_t)(t2 * 16 + r) * NPAD + c0;
            ABPack B;
            B.q[0] = *(const uint4*)bp;
            B.q[1] = *(const uint4*)(bp + 16);     // pad region: zeros
            C[t2] = wmma_bf16(A, B, C[t2]);
        }
    }

    // combine the 4 column-slice partials through LDS
    #pragma unroll
    for (int t2 = 0; t2 < 4; ++t2)
        #pragma unroll
        for (int v = 0; v < 8; ++v) lsC[w][t2][v][lane] = C[t2][v];
    lsL[w][lane] = lsum;
    __syncthreads();

    // wave w finalizes feature tile t = w
    float linv[8];
    #pragma unroll
    for (int v = 0; v < 8; ++v) {
        const int rr = hi * 8 + v;
        float l = 0.f;
        #pragma unroll
        for (int u = 0; u < 4; ++u) l += lsL[u][rr] + lsL[u][rr + 16];
        linv[v] = 1.0f / fmaxf(l, 1e-30f);
    }
    #pragma unroll
    for (int v = 0; v < 8; ++v) {
        float acc = 0.f;
        #pragma unroll
        for (int u = 0; u < 4; ++u) acc += lsC[u][w][v][lane];
        acc *= linv[v];
        float val = (acc > 0.f) ? acc : (__expf(acc) - 1.0f);   // elu
        out[(size_t)(row_base + hi * 8 + v) * OUTF + w * 16 + r] = val;
    }
}

// ---------------- launch ----------------
extern "C" void kernel_launch(void* const* d_in, const int* in_sizes, int n_in,
                              void* d_out, int out_size, void* d_ws, size_t ws_size,
                              hipStream_t stream) {
    const float* x        = (const float*)d_in[0];
    const float* adj      = (const float*)d_in[1];
    const float* Mm       = (const float*)d_in[2];
    const float* W        = (const float*)d_in[3];
    const float* a_self   = (const float*)d_in[4];
    const float* a_neighs = (const float*)d_in[5];
    float* out = (float*)d_out;

    // workspace layout (~1.43 MB)
    char* ws = (char*)d_ws;
    unsigned short* WT = (unsigned short*)(ws);                       //  64*512*2   = 64 KB
    unsigned short* hT = (unsigned short*)(ws + 65536);               //  64*10016*2 = 1.28 MB
    float* ssf = (float*)(ws + 65536 + (size_t)OUTF * NPAD * 2);      //  40 KB
    float* snb = ssf + NROWS;                                         //  40 KB
    (void)in_sizes; (void)n_in; (void)out_size; (void)ws_size;

    k_prep  <<<1, 256, 0, stream>>>(W, WT, hT);
    k_hidden<<<(625 + 7) / 8, 256, 0, stream>>>(x, WT, a_self, a_neighs, hT, ssf, snb);
    k_attn  <<<625, 128, 0, stream>>>(adj, Mm, ssf, snb, hT, out);
}
